// Attend_73796128080313
// MI455X (gfx1250) — matrix operand order repaired and checked by reading, not compiled, not run on hardware
//
#include <hip/hip_runtime.h>

#define LDK 72   // row stride in halves: 144B -> 36 dwords; 36*L mod 64 distinct for L=0..15

typedef __attribute__((ext_vector_type(16))) _Float16 v16h;
typedef __attribute__((ext_vector_type(8)))  _Float16 h8;
typedef __attribute__((ext_vector_type(4)))  _Float16 h4;
typedef __attribute__((ext_vector_type(8)))  float    v8f;
typedef __attribute__((ext_vector_type(4)))  float    f4;

__device__ __forceinline__ v16h make16(h8 lo, h8 hi) {
    v16h f;
#pragma unroll
    for (int i = 0; i < 8; ++i) { f[i] = lo[i]; f[i + 8] = hi[i]; }
    return f;
}

// A-operand (16x32 f16): lane half kbase in {0,8}; 16 contiguous K at k0 + 2*kbase
__device__ __forceinline__ v16h ld_afrag_lds(const _Float16* t, int row, int k0, int kbase) {
    const _Float16* p = t + row * LDK + k0 + 2 * kbase;
    return make16(*(const h8*)p, *(const h8*)(p + 8));
}

// B-operand (32x16 f16): lane half holds 16 contiguous K at k0 + 16*half
__device__ __forceinline__ v16h ld_bfrag_lds(const _Float16* t, int col, int k0, int khalf) {
    const _Float16* p = t + col * LDK + k0 + khalf;
    return make16(*(const h8*)p, *(const h8*)(p + 8));
}

// Q as B-operand (Q^T): lane = q row, 16 contiguous scaled d-values from global fp32
__device__ __forceinline__ v16h ld_qbfrag(const float* qr, int k0, int khalf, float scale) {
    const float* p = qr + k0 + khalf;
    f4 a = *(const f4*)(p);
    f4 b = *(const f4*)(p + 4);
    f4 c = *(const f4*)(p + 8);
    f4 d = *(const f4*)(p + 12);
    v16h f;
#pragma unroll
    for (int i = 0; i < 4; ++i) {
        f[i]      = (_Float16)(a[i] * scale);
        f[i + 4]  = (_Float16)(b[i] * scale);
        f[i + 8]  = (_Float16)(c[i] * scale);
        f[i + 12] = (_Float16)(d[i] * scale);
    }
    return f;
}

__device__ __forceinline__ float exp2_fast(float x) { return __builtin_amdgcn_exp2f(x); }

__global__ __launch_bounds__(128) void fa_fwd_kernel(const float* __restrict__ q,
                                                     const float* __restrict__ k,
                                                     const float* __restrict__ v,
                                                     float* __restrict__ out) {
    constexpr int N = 2048, D = 64;
    __shared__ _Float16 Ks[64 * LDK];        // K tile, row-major [kv_row][d]
    __shared__ _Float16 VTs[64 * LDK];       // V tile transposed [d][kv_row]
    __shared__ _Float16 PTs[4 * 16 * LDK];   // per-wave P^T tiles, row-major [q_row][kv_col]

    const int tid  = threadIdx.x;
    const int lane = tid & 31;
    const int wave = tid >> 5;
    const int half = lane >> 4;
    const int l16  = lane & 15;
    const int kbase = half * 8;
    const int khalf = half * 16;

    const int ntiles = N / 64;               // 32
    const int bh = blockIdx.x / ntiles;      // 0..31  (b*H + h)
    const int qt = blockIdx.x % ntiles;      // Q tile index
    const size_t base = (size_t)bh * N * D;
    // fold 1/sqrt(D) AND log2(e) into Q: S^T is computed directly in log2 domain,
    // so softmax uses raw v_exp_f32 (native exp2) with no per-element multiply.
    const float qscale = 0.125f * 1.44269504088896341f;

    // Q^T B-operand fragments (lane l16 = q row), held in registers for whole kernel
    const float* qr = q + base + (size_t)(qt * 64 + wave * 16 + l16) * D;
    const v16h bq0 = ld_qbfrag(qr, 0,  khalf, qscale);
    const v16h bq1 = ld_qbfrag(qr, 32, khalf, qscale);

    // per-lane softmax state (q row = l16, duplicated across wave halves)
    float m_i = -1e30f, l_i = 0.0f;
    v8f acc[4] = {};                         // O^T accumulators: tile nt -> d=16nt+8*half+r, q=l16

    _Float16* PTw = PTs + wave * 16 * LDK;

    for (int jt = 0; jt < ntiles; ++jt) {
        __syncthreads();                     // previous tile's K/V reads complete
        const float* kg = k + base + (size_t)jt * 64 * D;
        const float* vg = v + base + (size_t)jt * 64 * D;

        // K tile: each thread stages 8 contiguous cols of one row -> one b128 store
#pragma unroll
        for (int it = 0; it < 4; ++it) {
            int idx = tid + 128 * it;        // [0,512)
            int row = idx >> 3;              // 0..63
            int c8  = (idx & 7) << 3;        // 0..56
            f4 k0 = *(const f4*)(kg + row * D + c8);
            f4 k1 = *(const f4*)(kg + row * D + c8 + 4);
            h8 kh;
#pragma unroll
            for (int j = 0; j < 4; ++j) { kh[j] = (_Float16)k0[j]; kh[j + 4] = (_Float16)k1[j]; }
            *(h8*)(Ks + row * LDK + c8) = kh;           // ds_store_b128
        }

        // V tile transposed: 4 rows x 4 cols per thread -> packed b64 stores along kv
#pragma unroll
        for (int it = 0; it < 2; ++it) {
            int idx = tid + 128 * it;        // [0,256)
            int kv0 = (idx >> 4) << 2;       // 0..60 step 4
            int c4  = (idx & 15) << 2;       // 0..60
            f4 r0 = *(const f4*)(vg + (size_t)(kv0 + 0) * D + c4);
            f4 r1 = *(const f4*)(vg + (size_t)(kv0 + 1) * D + c4);
            f4 r2 = *(const f4*)(vg + (size_t)(kv0 + 2) * D + c4);
            f4 r3 = *(const f4*)(vg + (size_t)(kv0 + 3) * D + c4);
#pragma unroll
            for (int j = 0; j < 4; ++j) {
                h4 vh;
                vh[0] = (_Float16)r0[j];
                vh[1] = (_Float16)r1[j];
                vh[2] = (_Float16)r2[j];
                vh[3] = (_Float16)r3[j];
                *(h4*)(VTs + (c4 + j) * LDK + kv0) = vh; // ds_store_b64
            }
        }
        __syncthreads();

        if (jt + 1 < ntiles) {               // warm L2 for next tile during compute
            __builtin_prefetch(kg + 64 * D + tid * 8, 0, 0);
            __builtin_prefetch(vg + 64 * D + tid * 8, 0, 0);
        }

        // ---- S^T = K @ Q^T (log2 domain): tile t holds kv rows 16t..16t+15, q cols 0..15.
        //      C-layout: lane(half,l16) reg r <-> kv j = 16t + 8*half + r, q row = l16.
        v8f st[4];
#pragma unroll
        for (int t = 0; t < 4; ++t) {
            v8f c = {};
            v16h a0 = ld_afrag_lds(Ks, t * 16 + l16, 0, kbase);
            c = __builtin_amdgcn_wmma_f32_16x16x32_f16(false, a0, false, bq0, (short)0, c, false, false);
            v16h a1 = ld_afrag_lds(Ks, t * 16 + l16, 32, kbase);
            c = __builtin_amdgcn_wmma_f32_16x16x32_f16(false, a1, false, bq1, (short)0, c, false, false);
            st[t] = c;
        }

        // ---- online softmax over kv: FOUR independent max chains (ILP to cover
        //      WMMA->VALU hazard slots), each link a v_max3, then 3-op combine.
        float pm[4];
#pragma unroll
        for (int t = 0; t < 4; ++t) {
            float a = fmaxf(st[t][0], st[t][1]);
            a = fmaxf(fmaxf(a, st[t][2]), st[t][3]);
            a = fmaxf(fmaxf(a, st[t][4]), st[t][5]);
            a = fmaxf(fmaxf(a, st[t][6]), st[t][7]);
            pm[t] = a;
        }
        float tm = fmaxf(fmaxf(pm[0], pm[1]), fmaxf(pm[2], pm[3]));
        tm = fmaxf(tm, __shfl_xor(tm, 16));   // combine the two kv half-sets
        float mn = fmaxf(m_i, tm);
        float alpha = exp2_fast(m_i - mn);
        m_i = mn;

        // exponentiate in place (independent across all 32 values)
#pragma unroll
        for (int t = 0; t < 4; ++t)
#pragma unroll
            for (int r = 0; r < 8; ++r) st[t][r] = exp2_fast(st[t][r] - mn);

        // ---- P^T -> per-wave LDS immediately (stores in flight while we reduce):
        //      a lane's 8 regs are 8 contiguous kv cols -> one packed b128 store per tile
#pragma unroll
        for (int t = 0; t < 4; ++t) {
            h8 pk8;
#pragma unroll
            for (int r = 0; r < 8; ++r) pk8[r] = (_Float16)st[t][r];
            *(h8*)(PTw + l16 * LDK + t * 16 + half * 8) = pk8;   // ds_store_b128
        }

        // ---- row sum: independent vector adds then balanced horizontal tree
        v8f sv = (st[0] + st[1]) + (st[2] + st[3]);
        float ts = ((sv[0] + sv[1]) + (sv[2] + sv[3])) + ((sv[4] + sv[5]) + (sv[6] + sv[7]));
        ts += __shfl_xor(ts, 16);
        l_i = l_i * alpha + ts;

        // ---- rescale O^T accumulators (vector * scalar)
#pragma unroll
        for (int t = 0; t < 4; ++t) acc[t] = acc[t] * alpha;

        // P^T as B-operand: column = q row (l16), K = kv  (compiler inserts dscnt wait)
        v16h bp0 = ld_bfrag_lds(PTw, l16, 0,  khalf);
        v16h bp1 = ld_bfrag_lds(PTw, l16, 32, khalf);

        // ---- O^T += V^T @ P^T  (A-operand = V^T rows = head dims, from VTs)
#pragma unroll
        for (int nt = 0; nt < 4; ++nt) {
            v16h a0 = ld_afrag_lds(VTs, nt * 16 + l16, 0, kbase);
            acc[nt] = __builtin_amdgcn_wmma_f32_16x16x32_f16(false, a0, false, bp0, (short)0, acc[nt], false, false);
            v16h a1 = ld_afrag_lds(VTs, nt * 16 + l16, 32, kbase);
            acc[nt] = __builtin_amdgcn_wmma_f32_16x16x32_f16(false, a1, false, bp1, (short)0, acc[nt], false, false);
        }
    }

    // ---- epilogue: O = (O^T)^T / l ; lane owns q row l16, 8 contiguous d per tile
    float* orow = out + base + (size_t)(qt * 64 + wave * 16 + l16) * D;
    const float inv = 1.0f / l_i;
#pragma unroll
    for (int nt = 0; nt < 4; ++nt) {
        v8f o = acc[nt] * inv;               // packed muls
        f4 o0, o1;
#pragma unroll
        for (int i = 0; i < 4; ++i) { o0[i] = o[i]; o1[i] = o[i + 4]; }
        *(f4*)(orow + nt * 16 + half * 8)     = o0;
        *(f4*)(orow + nt * 16 + half * 8 + 4) = o1;
    }
}

extern "C" void kernel_launch(void* const* d_in, const int* in_sizes, int n_in,
                              void* d_out, int out_size, void* d_ws, size_t ws_size,
                              hipStream_t stream) {
    const float* q = (const float*)d_in[0];
    const float* k = (const float*)d_in[1];
    const float* v = (const float*)d_in[2];
    float* out = (float*)d_out;
    constexpr int B = 4, H = 8, N = 2048;
    dim3 grid(B * H * (N / 64));
    dim3 block(128);
    hipLaunchKernelGGL(fa_fwd_kernel, grid, block, 0, stream, q, k, v, out);
}
